// BaAttention_67078799229457
// MI455X (gfx1250) — matrix operand order repaired and checked by reading, not compiled, not run on hardware
//
#include <hip/hip_runtime.h>
#include <math.h>

// ---------------------------------------------------------------------------
// Bahdanau attention, MI455X (gfx1250, wave32, WMMA).
//
// Shapes: encoded [16,2048,1024] f32, decoder [16,1024] f32,
//         W1 [2048,2048] f32, w2 [2048] f32, out [16,1024] f32.
//
// Key algebra: S == 2H == 2048 makes the reference's raw reshape an index
// identity, so h[b,j,t] = (t<1024 ? encoded[b,j,t] : decoder[b,t-1024]).
// => t>=1024 columns collapse to rowsum(W1) * decoder scalar (no GEMM),
//    t< 1024 columns need W1 @ E_b : 16 x (2048x2048x1024) = 137 GFLOP.
// Footprint 151 MB fits the 192 MB L2 -> compute-bound -> WMMA bf16 with a
// 3-term hi/lo split (Ah*Bh + Ah*Bl + Al*Bh) for ~fp32 accuracy at bf16
// matrix-core rate. GEMM kernel is double-buffered in LDS: global loads for
// tile k+1 are issued into registers before the 24 WMMAs of tile k, and the
// convert+store lands in the alternate buffer with ONE barrier per K-step.
// All LDS staging stores are ds_store_b128 (8 bf16 per store).
// ---------------------------------------------------------------------------

#define NB   16      // batches
#define HD   1024    // hidden
#define S2   2048    // S == 2*HD
#define MT   128     // GEMM tile M (i index into W1 rows)
#define NT   128     // GEMM tile N (t index, 0..1023)
#define KT   32      // GEMM K step == WMMA K
#define LDA  40      // LDS row pitch (bf16 elems), padded: 80B stride
#define LDB  40
#define TSZ  (MT * LDA)          // one tile plane (== NT*LDB) in bf16 elems
#define NITER (S2 / KT)          // 64 K-steps

typedef __attribute__((ext_vector_type(16))) __bf16 v16bf;
typedef __attribute__((ext_vector_type(8)))  float  v8f;

__device__ __forceinline__ float fast_tanh(float x) {
#if __has_builtin(__builtin_amdgcn_tanhf)
  return __builtin_amdgcn_tanhf(x);        // V_TANH_F32 (gfx1250 trans op)
#else
  return tanhf(x);
#endif
}

__device__ __forceinline__ void bfsplit(float x, __bf16& h, __bf16& l) {
  h = (__bf16)x;
  l = (__bf16)(x - (float)h);
}

// A fragment (16x32 bf16, ISA 7.12.2): lane L: M=L&15, hf=L>>4.
// elem e <- K = 16*hf + e (this code's contiguous order, not the ISA table; matches B).
__device__ __forceinline__ v16bf load_frag_a(const __bf16* rowp, int hf) {
  union { v16bf v; uint4 q[2]; } u;
  u.q[0] = *(const uint4*)(rowp + 16 * hf);
  u.q[1] = *(const uint4*)(rowp + 16 * hf + 8);
  return u.v;
}

// B fragment (32x16 bf16): lane L: N=L&15, elem e <- K = 16*(L>>4) + e.
// LDS B tile is stored N-major (K contiguous) so this is 32 contiguous bytes.
__device__ __forceinline__ v16bf load_frag_b(const __bf16* p) {
  union { v16bf v; uint4 q[2]; } u;
  u.q[0] = ((const uint4*)p)[0];
  u.q[1] = ((const uint4*)p)[1];
  return u.v;
}

__device__ __forceinline__ v8f wmma_bf16(v16bf a, v16bf b, v8f c) {
  return __builtin_amdgcn_wmma_f32_16x16x32_bf16(
      false, a, false, b, (short)0, c, false, false);
}

// --- GEMM staging: each thread owns 8 K-contiguous elems of A and of B -----
// A: 128 rows x 32 k  -> 2 passes: thread p*2048+tid*8 (float4 x2 global read)
// B: 32 k x 128 t     -> 2 passes: thread reads a column of 8 k (stride HD)
__device__ __forceinline__ void stage_global(const float* __restrict__ Wbase,
                                             const float* __restrict__ Ebase,
                                             int k0, int tid,
                                             float4 ra[2][2], float rb[2][8]) {
#pragma unroll
  for (int p = 0; p < 2; ++p) {
    const int linear = p * 2048 + tid * 8;
    const int r = linear >> 5, c = linear & 31;
    const float* src = Wbase + (size_t)r * S2 + k0 + c;
    ra[p][0] = *(const float4*)(src);
    ra[p][1] = *(const float4*)(src + 4);
  }
#pragma unroll
  for (int p = 0; p < 2; ++p) {
    const int idx = p * 256 + tid;
    const int tt = idx & 127, kh = idx >> 7;       // kh in 0..3 (8 k each)
    const float* src = Ebase + (size_t)(k0 + kh * 8) * HD + tt;
#pragma unroll
    for (int j = 0; j < 8; ++j) rb[p][j] = src[(size_t)j * HD];
  }
}

__device__ __forceinline__ void store_lds(int tid,
                                          const float4 ra[2][2], const float rb[2][8],
                                          __bf16* Ah, __bf16* Al,
                                          __bf16* Bh, __bf16* Bl) {
#pragma unroll
  for (int p = 0; p < 2; ++p) {
    const int linear = p * 2048 + tid * 8;
    const int r = linear >> 5, c = linear & 31;
    const float* xs = (const float*)&ra[p][0];
    alignas(16) __bf16 h8[8], l8[8];
#pragma unroll
    for (int j = 0; j < 8; ++j) bfsplit(xs[j], h8[j], l8[j]);
    *(uint4*)(Ah + r * LDA + c) = *(const uint4*)h8;   // ds_store_b128
    *(uint4*)(Al + r * LDA + c) = *(const uint4*)l8;
  }
#pragma unroll
  for (int p = 0; p < 2; ++p) {
    const int idx = p * 256 + tid;
    const int tt = idx & 127, kh = idx >> 7;
    alignas(16) __bf16 h8[8], l8[8];
#pragma unroll
    for (int j = 0; j < 8; ++j) bfsplit(rb[p][j], h8[j], l8[j]);
    *(uint4*)(Bh + tt * LDB + kh * 8) = *(const uint4*)h8;
    *(uint4*)(Bl + tt * LDB + kh * 8) = *(const uint4*)l8;
  }
}

// ---------------------------------------------------------------------------
// Kernel 0: zero score[16*2048] and compute rowsum[i] = sum_j W1[i,j].
// grid 256 x block 256 (8 waves -> 8 rows per block).
// ---------------------------------------------------------------------------
__global__ __launch_bounds__(256)
void k0_prep(const float* __restrict__ W1, float* __restrict__ rowsum,
             float* __restrict__ score) {
  const int tid = threadIdx.x;
  const int gid = blockIdx.x * 256 + tid;
  if (gid < NB * S2) score[gid] = 0.f;

  const int lane = tid & 31, wave = tid >> 5;
  const int row  = blockIdx.x * 8 + wave;         // 256*8 = 2048 rows
  float s = 0.f;
  for (int c = lane; c < S2; c += 32) s += W1[(size_t)row * S2 + c];
#pragma unroll
  for (int off = 16; off; off >>= 1) s += __shfl_xor(s, off, 32);
  if (lane == 0) rowsum[row] = s;
}

// ---------------------------------------------------------------------------
// Kernel 1: fused GEMM + tanh + w2-dot for t in [0,1024).
//   score[b,t] += sum_i w2[i] * tanh( sum_s W1[i,s] * encoded[b,s,t] )
// 128x128 tile per block (256 threads = 8 waves, each wave 2x4 fragments),
// double-buffered LDS, one barrier per K-step, 24 WMMAs per wave per step.
// grid (8 t-tiles, 16 i-tiles, 16 batches).
// ---------------------------------------------------------------------------
__global__ __launch_bounds__(256)
void k1_gemm_score(const float* __restrict__ W1, const float* __restrict__ enc,
                   const float* __restrict__ w2, float* __restrict__ score) {
  const int b     = blockIdx.z;
  const int iBase = blockIdx.y * MT;
  const int tBase = blockIdx.x * NT;
  const int tid   = threadIdx.x;
  const int lane  = tid & 31, wave = tid >> 5;
  const int wm = wave >> 1, wn = wave & 1;        // wave -> 32-row x 64-col block
  const int hf = lane >> 4, ln16 = lane & 15;

  // double-buffered tile planes: [buf][Ah|Al|Bh|Bl], 80 KB total
  __shared__ __bf16 lds[2 * 4 * TSZ];
  __shared__ float  scoreBuf[NT];
  __shared__ float  w2s[MT];

  if (tid < NT) scoreBuf[tid] = 0.f;
  if (tid < MT) w2s[tid] = w2[iBase + tid];

  v8f acc[2][4];
#pragma unroll
  for (int mf = 0; mf < 2; ++mf)
#pragma unroll
    for (int nf = 0; nf < 4; ++nf)
#pragma unroll
      for (int v = 0; v < 8; ++v) acc[mf][nf][v] = 0.f;

  const float* Wbase = W1 + (size_t)iBase * S2;
  const float* Ebase = enc + (size_t)b * S2 * HD + tBase;

  float4 ra[2][2];
  float  rb[2][8];

  // prologue: stage + store tile 0 into buffer 0
  stage_global(Wbase, Ebase, 0, tid, ra, rb);
  store_lds(tid, ra, rb, lds, lds + TSZ, lds + 2 * TSZ, lds + 3 * TSZ);
  __syncthreads();

  for (int it = 0; it < NITER; ++it) {
    // issue next tile's global loads now; WMMAs below hide their latency
    if (it + 1 < NITER)
      stage_global(Wbase, Ebase, (it + 1) * KT, tid, ra, rb);

    __bf16* Ah = lds + (it & 1) * 4 * TSZ;
    __bf16* Al = Ah + TSZ;
    __bf16* Bh = Ah + 2 * TSZ;
    __bf16* Bl = Ah + 3 * TSZ;

    v16bf ah[2], al[2];
#pragma unroll
    for (int mf = 0; mf < 2; ++mf) {
      const int row = 32 * wm + 16 * mf + ln16;
      ah[mf] = load_frag_a(Ah + row * LDA, hf);
      al[mf] = load_frag_a(Al + row * LDA, hf);
    }
#pragma unroll
    for (int nf = 0; nf < 4; ++nf) {
      const int col = 64 * wn + 16 * nf + ln16;
      const v16bf bh = load_frag_b(Bh + col * LDB + 16 * hf);
      const v16bf bl = load_frag_b(Bl + col * LDB + 16 * hf);
#pragma unroll
      for (int mf = 0; mf < 2; ++mf) {
        acc[mf][nf] = wmma_bf16(ah[mf], bh, acc[mf][nf]);  // hi*hi
        acc[mf][nf] = wmma_bf16(ah[mf], bl, acc[mf][nf]);  // hi*lo
        acc[mf][nf] = wmma_bf16(al[mf], bh, acc[mf][nf]);  // lo*hi
      }
    }

    // convert + store next tile into the alternate buffer; single barrier.
    // Safe: the buffer written here was last read at iter it-1, strictly
    // before that iteration's closing barrier.
    if (it + 1 < NITER) {
      __bf16* nAh = lds + ((it + 1) & 1) * 4 * TSZ;
      store_lds(tid, ra, rb, nAh, nAh + TSZ, nAh + 2 * TSZ, nAh + 3 * TSZ);
      __syncthreads();
    }
  }

  // --- epilogue: tanh, scale by w2[i], reduce over i, accumulate score[b,t].
  // C/D layout: elem v at lane L -> M = v + 8*(L>>4), N = L&15.
#pragma unroll
  for (int mf = 0; mf < 2; ++mf) {
#pragma unroll
    for (int nf = 0; nf < 4; ++nf) {
      float part = 0.f;
#pragma unroll
      for (int v = 0; v < 8; ++v) {
        const int il = 32 * wm + 16 * mf + 8 * hf + v;
        part += w2s[il] * fast_tanh(acc[mf][nf][v]);
      }
      part += __shfl_xor(part, 16, 32);             // fold the two M-halves
      if (hf == 0)
        atomicAdd(&scoreBuf[64 * wn + 16 * nf + ln16], part);
    }
  }
  __syncthreads();
  if (tid < NT)
    atomicAdd(&score[(size_t)b * S2 + tBase + tid], scoreBuf[tid]);
}

// ---------------------------------------------------------------------------
// Kernel 2: closed-form scores for t in [1024,2048):
//   score[b,t] = sum_i w2[i] * tanh( decoder[b,t-1024] * rowsum[i] )
// One wave per (b,t); 8 waves per block -> 2048 blocks.
// ---------------------------------------------------------------------------
__global__ __launch_bounds__(256)
void k2_tail(const float* __restrict__ dec, const float* __restrict__ rowsum,
             const float* __restrict__ w2, float* __restrict__ score) {
  const int wid  = (blockIdx.x * 256 + threadIdx.x) >> 5;   // 0..16383
  const int lane = threadIdx.x & 31;
  const int b = wid >> 10, tt = wid & 1023;
  const float d = dec[(size_t)b * HD + tt];
  float acc = 0.f;
  for (int i = lane; i < S2; i += 32)
    acc += w2[i] * fast_tanh(d * rowsum[i]);
#pragma unroll
  for (int off = 16; off; off >>= 1) acc += __shfl_xor(acc, off, 32);
  if (lane == 0) score[(size_t)b * S2 + HD + tt] = acc;
}

// ---------------------------------------------------------------------------
// Kernel 3: softmax over score[b,:] + weighted sum over encoded -> out[b,:].
// grid (4 feature-chunks of 256, 16 batches); one HBM pass over encoded.
// ---------------------------------------------------------------------------
__global__ __launch_bounds__(256)
void k3_softmax_out(const float* __restrict__ enc, const float* __restrict__ score,
                    float* __restrict__ out) {
  __shared__ float ls[S2];
  __shared__ float red[256];
  const int b = blockIdx.y, chunk = blockIdx.x, tid = threadIdx.x;

  for (int j = tid; j < S2; j += 256) ls[j] = score[(size_t)b * S2 + j];
  __syncthreads();

  float m = -INFINITY;
  for (int j = tid; j < S2; j += 256) m = fmaxf(m, ls[j]);
  red[tid] = m; __syncthreads();
  for (int off = 128; off; off >>= 1) {
    if (tid < off) red[tid] = fmaxf(red[tid], red[tid + off]);
    __syncthreads();
  }
  m = red[0]; __syncthreads();

  float sum = 0.f;
  for (int j = tid; j < S2; j += 256) {
    const float e = __expf(ls[j] - m);
    ls[j] = e;
    sum += e;
  }
  red[tid] = sum; __syncthreads();
  for (int off = 128; off; off >>= 1) {
    if (tid < off) red[tid] += red[tid + off];
    __syncthreads();
  }
  const float inv = 1.f / red[0];

  const int f = chunk * 256 + tid;
  const float* ep = enc + (size_t)b * S2 * HD + f;
  float acc = 0.f;
#pragma unroll 4
  for (int s = 0; s < S2; ++s) acc += ls[s] * ep[(size_t)s * HD];
  out[(size_t)b * HD + f] = acc * inv;
}

// ---------------------------------------------------------------------------
extern "C" void kernel_launch(void* const* d_in, const int* in_sizes, int n_in,
                              void* d_out, int out_size, void* d_ws, size_t ws_size,
                              hipStream_t stream) {
  const float* enc = (const float*)d_in[0];   // [16,2048,1024]
  const float* dec = (const float*)d_in[1];   // [16,1024]
  const float* W1  = (const float*)d_in[2];   // [2048,2048]
  const float* w2  = (const float*)d_in[3];   // [2048]
  float* out = (float*)d_out;                 // [16,1024]

  // workspace: score (16*2048 f32) + rowsum (2048 f32) = 136 KB
  float* score  = (float*)d_ws;
  float* rowsum = score + NB * S2;

  k0_prep<<<256, 256, 0, stream>>>(W1, rowsum, score);
  k1_gemm_score<<<dim3(HD / NT, S2 / MT, NB), 256, 0, stream>>>(W1, enc, w2, score);
  k2_tail<<<(NB * HD) / 8, 256, 0, stream>>>(dec, rowsum, w2, score);
  k3_softmax_out<<<dim3(HD / 256, NB), 256, 0, stream>>>(enc, score, out);
}
